// DCD_52450140619545
// MI455X (gfx1250) — matrix operand order repaired and checked by reading, not compiled, not run on hardware
//
#include <hip/hip_runtime.h>
#include <hip/hip_bf16.h>

typedef __attribute__((ext_vector_type(16))) __bf16 v16bf;
typedef __attribute__((ext_vector_type(8)))  float  v8f;

#define B_   4
#define C_   64
#define H_   128
#define W_   128
#define K_   9
#define KD   576      // C_ * K_
#define KSTEPS 18     // KD / 32
#define TILE 16

__device__ __forceinline__ unsigned short f2bf(float f) {
  unsigned u = __float_as_uint(f);
  u += 0x7FFFu + ((u >> 16) & 1u);   // round-to-nearest-even
  return (unsigned short)(u >> 16);
}

// ---------------------------------------------------------------- prep weights
__global__ __launch_bounds__(256) void prep_weights_kernel(
    const float* __restrict__ w_off, const float* __restrict__ w_align,
    unsigned short* __restrict__ woff_bf, unsigned short* __restrict__ walign_bf)
{
  int i = blockIdx.x * blockDim.x + threadIdx.x;
  if (i < 32 * KD) {                       // pad 27 -> 32 rows with zeros
    int o = i / KD;
    int k = i - o * KD;
    woff_bf[i] = f2bf(o < 27 ? w_off[o * KD + k] : 0.0f);
  } else {
    int j = i - 32 * KD;
    if (j < 64 * KD) walign_bf[j] = f2bf(w_align[j]);
  }
}

// --------------------------------------------------------------- WMMA fragments
struct Frag {
  union { v16bf v; unsigned u[8]; uint4 q[2]; };
};

__device__ __forceinline__ void load_a(Frag& A, const unsigned short* arow,
                                       int ks, int krow) {
#pragma unroll
  for (int t = 0; t < 8; ++t) {            // ISA 16-bit A layout: pairs of K
    int k = ks * 32 + ((t < 4) ? 0 : 16) + krow + 2 * (t & 3);
    A.u[t] = *(const unsigned*)(arow + k);
  }
}

__device__ __forceinline__ void load_b(Frag& B, const unsigned short* brow,
                                       int ks, int kbo) {
  const unsigned short* bp = brow + ks * 32 + kbo;   // K kbo..kbo+7, 16+kbo..16+kbo+7
  B.q[0] = *(const uint4*)(bp);
  B.q[1] = *(const uint4*)(bp + 16);
}

// 16xKD (bf16) x KDx16 (bf16) -> 16x16 f32, 18 k-steps, double-buffered so the
// wait before each WMMA covers loads issued one iteration earlier (latency hide).
__device__ __forceinline__ v8f gemm16xKD(const unsigned short* a_lds,
                                         const unsigned short* b_row,
                                         int lane)
{
  v8f acc = {0.f, 0.f, 0.f, 0.f, 0.f, 0.f, 0.f, 0.f};
  const int mA   = lane & 15;
  const int krow = (lane < 16) ? 0 : 8;    // A: upper half-wave holds K+8
  const int kbo  = (lane < 16) ? 0 : 8;    // B: upper half-wave holds K+8
  const unsigned short* arow = a_lds + mA * KD;

  Frag A[2], Bf[2];
  load_a(A[0], arow, 0, krow);
  load_b(Bf[0], b_row, 0, kbo);
#pragma unroll
  for (int ks = 0; ks < KSTEPS; ++ks) {
    const int cur = ks & 1;
    const int nxt = cur ^ 1;
    if (ks + 1 < KSTEPS) {                 // prefetch next step's fragments
      load_a(A[nxt], arow, ks + 1, krow);
      load_b(Bf[nxt], b_row, ks + 1, kbo);
    }
    acc = __builtin_amdgcn_wmma_f32_16x16x32_bf16(false, A[cur].v, false,
                                                  Bf[cur].v, (short)0, acc,
                                                  false, false);
  }
  return acc;
}

// ---------------------------------------------------- fused deformable pipeline
__global__ __launch_bounds__(128) void fused_dcn_kernel(
    const float* __restrict__ x,
    const float* __restrict__ b_off,
    const float* __restrict__ b_align,
    const unsigned short* __restrict__ woff_bf,    // [32][576]
    const unsigned short* __restrict__ walign_bf,  // [64][576]
    float* __restrict__ wfield)
{
  __shared__ __align__(16) unsigned short a_p[TILE * KD]; // im2col patch
  __shared__ __align__(16) unsigned short a_s[TILE * KD]; // sampled matrix
  __shared__ float off_lds[27 * TILE];
  __shared__ float cwgt[144 * 4];
  __shared__ int   cidx[144 * 4];

  const int tid  = threadIdx.x;
  const int wave = tid >> 5;
  const int lane = tid & 31;

  const int tileid = blockIdx.x;          // b * 128 rows * 8 xtiles
  const int xt = tileid & 7;
  const int y  = (tileid >> 3) & 127;
  const int b  = tileid >> 10;
  const int x0 = xt * TILE;

  // ---- Phase 1: im2col (zero-padded) -> a_p as bf16
  for (int idx = tid; idx < 144 * C_; idx += 128) {
    int pt = idx % 144;                   // pt = kk*16 + m (coalesced in m)
    int c  = idx / 144;
    int kk = pt >> 4;
    int m  = pt & 15;
    int yy = y + (kk / 3) - 1;
    int xx = x0 + m + (kk % 3) - 1;
    float v = 0.0f;
    if (yy >= 0 && yy < H_ && xx >= 0 && xx < W_)
      v = x[((b * C_ + c) * H_ + yy) * W_ + xx];
    a_p[m * KD + c * K_ + kk] = f2bf(v);
  }
  __syncthreads();

  // ---- Phase 2: offset conv (27 ch, padded 32) via WMMA on waves 0..1
  if (wave < 2) {
    const unsigned short* brow = woff_bf + (wave * 16 + (lane & 15)) * KD;
    v8f acc = gemm16xKD(a_p, brow, lane);
    int n = wave * 16 + (lane & 15);
    if (n < 27) {
      float bb = b_off[n];
      int mbase = (lane < 16) ? 0 : 8;    // C/D layout: M = r + 8*(lane>=16)
#pragma unroll
      for (int r = 0; r < 8; ++r)
        off_lds[n * TILE + (mbase + r)] = acc[r] + bb;
    }
  }
  __syncthreads();

  // ---- Phase 3: per (tap, pixel) bilinear corner weights (mask folded in)
  if (tid < 144) {
    int kk = tid >> 4;
    int m  = tid & 15;
    float dy = off_lds[(2 * kk) * TILE + m];
    float dx = off_lds[(2 * kk + 1) * TILE + m];
    float ml = off_lds[(18 + kk) * TILE + m];
    float mask = 2.0f / (1.0f + __expf(-ml));
    float ysf = (float)y + (float)(kk / 3 - 1) + dy;
    float xsf = (float)(x0 + m) + (float)(kk % 3 - 1) + dx;
    float yf = floorf(ysf), xf = floorf(xsf);
    float fy = ysf - yf, fx = xsf - xf;
    int yi = (int)yf, xi = (int)xf;
#pragma unroll
    for (int cr = 0; cr < 4; ++cr) {
      int cy = cr >> 1, cx = cr & 1;
      int yc = yi + cy, xc = xi + cx;
      float wgt = (cy ? fy : (1.0f - fy)) * (cx ? fx : (1.0f - fx));
      bool valid = (yc >= 0) && (yc < H_) && (xc >= 0) && (xc < W_);
      int ycl = yc < 0 ? 0 : (yc > H_ - 1 ? H_ - 1 : yc);
      int xcl = xc < 0 ? 0 : (xc > W_ - 1 ? W_ - 1 : xc);
      cwgt[tid * 4 + cr] = valid ? (mask * wgt) : 0.0f;
      cidx[tid * 4 + cr] = ycl * W_ + xcl;
    }
  }
  __syncthreads();

  // ---- Phase 4: gather-build sampled matrix -> a_s (bf16)
  for (int idx = tid; idx < 144 * C_; idx += 128) {
    int pt = idx % 144;
    int c  = idx / 144;
    int kk = pt >> 4;
    int m  = pt & 15;
    const float* xp = x + (b * C_ + c) * (H_ * W_);
    float v = cwgt[pt * 4 + 0] * xp[cidx[pt * 4 + 0]]
            + cwgt[pt * 4 + 1] * xp[cidx[pt * 4 + 1]]
            + cwgt[pt * 4 + 2] * xp[cidx[pt * 4 + 2]]
            + cwgt[pt * 4 + 3] * xp[cidx[pt * 4 + 3]];
    a_s[m * KD + c * K_ + kk] = f2bf(v);
  }
  __syncthreads();

  // ---- Phase 5: align GEMM via WMMA (4 waves x 16 out ch), act, store
  {
    const unsigned short* brow = walign_bf + (wave * 16 + (lane & 15)) * KD;
    v8f acc = gemm16xKD(a_s, brow, lane);
    int n = wave * 16 + (lane & 15);
    float bb = b_align[n];
    int mbase = (lane < 16) ? 0 : 8;
    float* wp = wfield + ((b * C_ + n) * H_ + y) * W_ + x0;
#pragma unroll
    for (int r = 0; r < 8; ++r) {
      float v = acc[r] + bb;
      float sg = 1.0f / (1.0f + __expf(-v));
      wp[mbase + r] = __expf(sg);
    }
  }
}

// ------------------------------------------------------- pooled ratio (÷9 cancels)
__global__ __launch_bounds__(256) void pool_div_kernel(
    const float* __restrict__ x, const float* __restrict__ wfield,
    float* __restrict__ out)
{
  int i = blockIdx.x * blockDim.x + threadIdx.x;
  if (i >= B_ * C_ * 64 * 64) return;
  int ox = i & 63;
  int oy = (i >> 6) & 63;
  int bc = i >> 12;                        // b*C_ + c
  const float* xp = x + bc * (H_ * W_);
  const float* wp = wfield + bc * (H_ * W_);
  float num = 0.0f, den = 0.0f;
#pragma unroll
  for (int ky = 0; ky < 3; ++ky) {
    int yy = 2 * oy + ky - 1;
    if (yy < 0 || yy >= H_) continue;
#pragma unroll
    for (int kx = 0; kx < 3; ++kx) {
      int xx = 2 * ox + kx - 1;
      if (xx < 0 || xx >= W_) continue;
      float wv = wp[yy * W_ + xx];
      num += wv * xp[yy * W_ + xx];
      den += wv;
    }
  }
  out[i] = num / den;
}

extern "C" void kernel_launch(void* const* d_in, const int* in_sizes, int n_in,
                              void* d_out, int out_size, void* d_ws, size_t ws_size,
                              hipStream_t stream)
{
  const float* x       = (const float*)d_in[0];
  const float* w_off   = (const float*)d_in[1];
  const float* b_off   = (const float*)d_in[2];
  const float* w_align = (const float*)d_in[3];
  const float* b_align = (const float*)d_in[4];
  float* out = (float*)d_out;

  // Workspace layout:
  //   [0, 16777216)                 : wfield (4*64*128*128 f32)
  //   [16777216, +36864)            : w_off bf16 padded [32][576]
  //   [16814080, +73728)            : w_align bf16 [64][576]
  char* ws = (char*)d_ws;
  float* wfield             = (float*)ws;
  unsigned short* woff_bf   = (unsigned short*)(ws + 16777216);
  unsigned short* walign_bf = (unsigned short*)(ws + 16777216 + 36864);

  int prep_n = 32 * KD + 64 * KD;  // 55296
  prep_weights_kernel<<<(prep_n + 255) / 256, 256, 0, stream>>>(
      w_off, w_align, woff_bf, walign_bf);

  fused_dcn_kernel<<<B_ * H_ * (W_ / TILE), 128, 0, stream>>>(
      x, b_off, b_align, woff_bf, walign_bf, wfield);

  pool_div_kernel<<<(B_ * C_ * 64 * 64 + 255) / 256, 256, 0, stream>>>(
      x, wfield, out);
}
